// NNConvNet_64089501991007
// MI455X (gfx1250) — matrix operand order repaired and checked by reading, not compiled, not run on hardware
//
#include <hip/hip_runtime.h>
#include <hip/hip_bf16.h>
#include <math.h>

#define NNODES 16384
#define NEDGES 131072
#define EDGE_DIM 16
#define IN_CH 32
#define OUT_CH 32
#define HID_EDGE 64
#define N_GRAPHS 64

typedef __attribute__((ext_vector_type(16))) _Float16 v16h;
typedef __attribute__((ext_vector_type(8)))  float    v8f;

// ---------------------------------------------------------------------------
// WMMA helper: D = A(16x32 f16) * B(32x16 f16) + C(16x16 f32)
// ---------------------------------------------------------------------------
__device__ __forceinline__ v8f wmma32(v16h a, v16h b, v8f c) {
    return __builtin_amdgcn_wmma_f32_16x16x32_f16(
        /*neg_a=*/false, a, /*neg_b=*/false, b,
        /*c_mod=*/(short)0, c, /*reuse_a=*/false, /*reuse_b=*/false);
}

// A-fragment load: one 32-float row -> 16xf16 per lane, ISA 16-bit A layout:
// lane<16: K = {0..7, 16..23};  lane>=16: K = {8..15, 24..31}
__device__ __forceinline__ v16h load_row32_afrag(const float* __restrict__ rowp, int half_) {
    float4 a0 = *(const float4*)(rowp + half_ * 8);
    float4 a1 = *(const float4*)(rowp + half_ * 8 + 4);
    float4 a2 = *(const float4*)(rowp + 16 + half_ * 8);
    float4 a3 = *(const float4*)(rowp + 16 + half_ * 8 + 4);
    v16h v;
    v[0]=(_Float16)a0.x;  v[1]=(_Float16)a0.y;  v[2]=(_Float16)a0.z;  v[3]=(_Float16)a0.w;
    v[4]=(_Float16)a1.x;  v[5]=(_Float16)a1.y;  v[6]=(_Float16)a1.z;  v[7]=(_Float16)a1.w;
    v[8]=(_Float16)a2.x;  v[9]=(_Float16)a2.y;  v[10]=(_Float16)a2.z; v[11]=(_Float16)a2.w;
    v[12]=(_Float16)a3.x; v[13]=(_Float16)a3.y; v[14]=(_Float16)a3.z; v[15]=(_Float16)a3.w;
    return v;
}

// ---------------------------------------------------------------------------
// Pack a row-major f32 matrix W[K x Ncols] into B-fragment-linear f16:
// out[((t*nKT + kt)*32 + lane)*16 + j] = W[kt*32 + (lane>>4)*8 + (j<8 ? j : j+8)][t*16 + (lane&15)]
// (ISA 16-bit B layout: lanes 0-15 hold K={0..7,16..23}, lanes 16-31 hold K={8..15,24..31})
// Rows >= Krows are zero padding.
// ---------------------------------------------------------------------------
__global__ void pack_b_kernel(const float* __restrict__ W, _Float16* __restrict__ out,
                              int Krows, int Ncols, int nKT, int nNT) {
    int tid = blockIdx.x * blockDim.x + threadIdx.x;
    int total = nNT * nKT * 32;
    if (tid >= total) return;
    int lane = tid & 31;
    int kt = (tid >> 5) % nKT;
    int t  = (tid >> 5) / nKT;
    _Float16* dst = out + ((size_t)(t * nKT + kt) * 32 + lane) * 16;
    int n = t * 16 + (lane & 15);
    int kbase = kt * 32 + (lane >> 4) * 8;
    for (int j = 0; j < 16; ++j) {
        int k = kbase + (j < 8 ? j : j + 8);
        float v = (k < Krows) ? W[(size_t)k * Ncols + n] : 0.f;
        dst[j] = (_Float16)v;
    }
}

__global__ void zero_kernel(float* __restrict__ p, int n) {
    int t = blockIdx.x * blockDim.x + threadIdx.x;
    if (t < n) p[t] = 0.f;
}

// in-degree per node + nodes per graph
__global__ void degree_kernel(const long long* __restrict__ ei,
                              const long long* __restrict__ batch,
                              float* __restrict__ cnt, float* __restrict__ gcnt) {
    int t = blockIdx.x * blockDim.x + threadIdx.x;
    if (t < NEDGES) atomicAdd(&cnt[(int)ei[NEDGES + t]], 1.0f);
    if (t < NNODES) atomicAdd(&gcnt[(int)batch[t]], 1.0f);
}

// ---------------------------------------------------------------------------
// Fused NNConv message kernel. One wave == one 16-edge tile.
//   h   = relu(edge_attr @ w1 + b1)                       (4 WMMA, K padded to 32)
//   msg = sum_h ( (h[:,h] * x_src) @ w2_flat[h*32:+32,:] ) (128 WMMA)
//       + x_src @ reshape(b2,[32,32])                      (2 WMMA)
//   agg[dst] += msg                                        (atomic scatter)
// ---------------------------------------------------------------------------
__global__ void __launch_bounds__(128) nnconv_msg_kernel(
    const float* __restrict__ x,          // [NNODES, 32]
    const float* __restrict__ ea,         // [NEDGES, 16]
    const long long* __restrict__ ei,     // [2, NEDGES]
    const _Float16* __restrict__ w1B,     // packed [4][1][32][16]
    const float* __restrict__ b1,         // [64]
    const _Float16* __restrict__ w2B,     // packed [2][64][32][16]
    const _Float16* __restrict__ b2B,     // packed [2][1][32][16]
    float* __restrict__ agg)              // [NNODES, 32] accumulator
{
    __shared__ _Float16 hbuf[4][16 * 64];   // per-wave h tile, row-major [row][64]
    const int lane  = threadIdx.x & 31;
    const int wv    = threadIdx.x >> 5;
    const int half_ = lane >> 4;
    const int nn    = lane & 15;
    const int e0    = (blockIdx.x * 4 + wv) * 16;

    // ---- stage 1: edge MLP hidden, via WMMA (A = edge_attr padded 16->32 K) ----
    v16h af = {};
    {
        const float* rp = ea + (size_t)(e0 + nn) * EDGE_DIM + half_ * 8;
        float4 f0 = *(const float4*)rp;
        float4 f1 = *(const float4*)(rp + 4);
        af[0]=(_Float16)f0.x; af[1]=(_Float16)f0.y; af[2]=(_Float16)f0.z; af[3]=(_Float16)f0.w;
        af[4]=(_Float16)f1.x; af[5]=(_Float16)f1.y; af[6]=(_Float16)f1.z; af[7]=(_Float16)f1.w;
        // elements 8..15 correspond to K>=16 -> stay zero (EDGE_DIM==16)
    }
#pragma unroll
    for (int t = 0; t < 4; ++t) {
        v16h bf = *(const v16h*)(w1B + (size_t)(t * 32 + lane) * 16);
        v8f c = {};
        c = wmma32(af, bf, c);
        int col = t * 16 + nn;
        float bb = b1[col];
#pragma unroll
        for (int r = 0; r < 8; ++r) {
            float v = c[r] + bb;
            hbuf[wv][(r + 8 * half_) * 64 + col] = (_Float16)(v > 0.f ? v : 0.f);
        }
    }
    asm volatile("s_wait_dscnt 0" ::: "memory");   // LDS writes visible to own wave's reads

    // ---- gather source node features as A fragment ----
    const long long srcn = ei[e0 + nn];
    v16h xs = load_row32_afrag(x + (size_t)srcn * 32, half_);

    // ---- stage 2: msg = Z @ w2_flat ; A-chunk(h) = xs scaled by h_row[h] ----
    v8f acc0 = {}, acc1 = {};
    const _Float16* hrow = &hbuf[wv][nn * 64];
#pragma unroll 4
    for (int h = 0; h < HID_EDGE; ++h) {
        _Float16 s = hrow[h];
        v16h ah;
#pragma unroll
        for (int j = 0; j < 16; ++j) ah[j] = xs[j] * s;
        v16h bf0 = *(const v16h*)(w2B + (size_t)(h * 32 + lane) * 16);
        v16h bf1 = *(const v16h*)(w2B + (size_t)((HID_EDGE + h) * 32 + lane) * 16);
        acc0 = wmma32(ah, bf0, acc0);
        acc1 = wmma32(ah, bf1, acc1);
    }
    // ---- b2 bias term: xs @ reshape(b2, [32,32]) ----
    {
        v16h bb0 = *(const v16h*)(b2B + (size_t)lane * 16);
        v16h bb1 = *(const v16h*)(b2B + (size_t)(32 + lane) * 16);
        acc0 = wmma32(xs, bb0, acc0);
        acc1 = wmma32(xs, bb1, acc1);
    }

    // ---- scatter-add messages into agg[dst] ----
    const long long* dstp = ei + NEDGES;
#pragma unroll
    for (int r = 0; r < 8; ++r) {
        int erow = e0 + r + 8 * half_;                 // C-layout row of this element
        float* base = agg + (size_t)dstp[erow] * 32;
        atomicAdd(base + nn,      acc0[r]);            // cols  0..15
        atomicAdd(base + 16 + nn, acc1[r]);            // cols 16..31
    }
}

// ---------------------------------------------------------------------------
// Node update: y = agg/max(cnt,1) + x @ root + bias  (WMMA for x@root),
// plus per-channel BN statistics accumulation.
// ---------------------------------------------------------------------------
__global__ void __launch_bounds__(128) node_update_kernel(
    const float* __restrict__ x, const _Float16* __restrict__ rootB,
    const float* __restrict__ bias, const float* __restrict__ agg,
    const float* __restrict__ cnt, float* __restrict__ y,
    float* __restrict__ bnsum, float* __restrict__ bnsq)
{
    const int lane  = threadIdx.x & 31;
    const int wv    = threadIdx.x >> 5;
    const int half_ = lane >> 4;
    const int nn    = lane & 15;
    const int n0    = (blockIdx.x * 4 + wv) * 16;

    v16h xs = load_row32_afrag(x + (size_t)(n0 + nn) * 32, half_);
    v16h r0 = *(const v16h*)(rootB + (size_t)lane * 16);
    v16h r1 = *(const v16h*)(rootB + (size_t)(32 + lane) * 16);
    v8f c0 = {}, c1 = {};
    c0 = wmma32(xs, r0, c0);
    c1 = wmma32(xs, r1, c1);

    float b0v = bias[nn], b1v = bias[16 + nn];
    float s0 = 0.f, q0 = 0.f, s1 = 0.f, q1 = 0.f;
#pragma unroll
    for (int r = 0; r < 8; ++r) {
        int n = n0 + r + 8 * half_;
        float cn = cnt[n];
        float inv = 1.f / (cn > 1.f ? cn : 1.f);
        float v0 = c0[r] + agg[(size_t)n * 32 + nn] * inv + b0v;
        float v1 = c1[r] + agg[(size_t)n * 32 + 16 + nn] * inv + b1v;
        y[(size_t)n * 32 + nn]      = v0;
        y[(size_t)n * 32 + 16 + nn] = v1;
        s0 += v0; q0 += v0 * v0;
        s1 += v1; q1 += v1 * v1;
    }
    atomicAdd(&bnsum[nn],      s0); atomicAdd(&bnsq[nn],      q0);
    atomicAdd(&bnsum[16 + nn], s1); atomicAdd(&bnsq[16 + nn], q1);
}

__global__ void bn_relu_kernel(const float* __restrict__ y,
                               const float* __restrict__ bnsum, const float* __restrict__ bnsq,
                               const float* __restrict__ gamma, const float* __restrict__ beta,
                               float* __restrict__ outp) {
    int t = blockIdx.x * blockDim.x + threadIdx.x;
    if (t >= NNODES * 32) return;
    int c = t & 31;
    const float invN = 1.0f / (float)NNODES;
    float m   = bnsum[c] * invN;
    float var = bnsq[c] * invN - m * m;
    float v = (y[t] - m) * rsqrtf(var + 1e-5f) * gamma[c] + beta[c];
    outp[t] = v > 0.f ? v : 0.f;
}

__global__ void pool_kernel(const float* __restrict__ feat, const long long* __restrict__ batch,
                            float* __restrict__ pooled) {
    int t = blockIdx.x * blockDim.x + threadIdx.x;
    if (t >= NNODES * 32) return;
    int n = t >> 5, c = t & 31;
    atomicAdd(&pooled[(size_t)batch[n] * 32 + c], feat[t]);
}

__global__ void final_mlp_kernel(const float* __restrict__ pooled, const float* __restrict__ gcnt,
                                 const float* __restrict__ w1, const float* __restrict__ b1,
                                 const float* __restrict__ w2, const float* __restrict__ b2,
                                 float* __restrict__ out) {
    __shared__ float hid[N_GRAPHS * 64];
    for (int i = threadIdx.x; i < N_GRAPHS * 64; i += blockDim.x) {
        int g = i >> 6, j = i & 63;
        float inv = 1.f / fmaxf(gcnt[g], 1.f);
        float acc = b1[j];
        for (int c = 0; c < 32; ++c) acc += pooled[g * 32 + c] * inv * w1[c * 64 + j];
        hid[i] = acc > 0.f ? acc : 0.f;
    }
    __syncthreads();
    for (int i = threadIdx.x; i < N_GRAPHS * 10; i += blockDim.x) {
        int g = i / 10, j = i % 10;
        float acc = b2[j];
        for (int c = 0; c < 64; ++c) acc += hid[g * 64 + c] * w2[c * 10 + j];
        out[i] = acc;
    }
}

// ---------------------------------------------------------------------------
extern "C" void kernel_launch(void* const* d_in, const int* in_sizes, int n_in,
                              void* d_out, int out_size, void* d_ws, size_t ws_size,
                              hipStream_t stream) {
    (void)in_sizes; (void)n_in; (void)out_size; (void)ws_size;
    const float*     x         = (const float*)d_in[0];
    const float*     edge_attr = (const float*)d_in[1];
    const long long* ei        = (const long long*)d_in[2];
    const long long* batch     = (const long long*)d_in[3];
    const float* w1_0 = (const float*)d_in[4];  const float* b1_0 = (const float*)d_in[5];
    const float* w2_0 = (const float*)d_in[6];  const float* b2_0 = (const float*)d_in[7];
    const float* root_0=(const float*)d_in[8];  const float* bias_0=(const float*)d_in[9];
    const float* gamma_0=(const float*)d_in[10];const float* beta_0=(const float*)d_in[11];
    const float* w1_1 = (const float*)d_in[12]; const float* b1_1 = (const float*)d_in[13];
    const float* w2_1 = (const float*)d_in[14]; const float* b2_1 = (const float*)d_in[15];
    const float* root_1=(const float*)d_in[16]; const float* bias_1=(const float*)d_in[17];
    const float* gamma_1=(const float*)d_in[18];const float* beta_1=(const float*)d_in[19];
    const float* mlp_w1=(const float*)d_in[20]; const float* mlp_b1=(const float*)d_in[21];
    const float* mlp_w2=(const float*)d_in[22]; const float* mlp_b2=(const float*)d_in[23];
    float* out = (float*)d_out;

    // ---- workspace carve-up (256B aligned sections) ----
    char* ws = (char*)d_ws;
    size_t off = 0;
    auto alloc = [&](size_t bytes) -> void* {
        void* p = ws + off;
        off += (bytes + 255) & ~(size_t)255;
        return p;
    };
    float* agg    = (float*)alloc((size_t)NNODES * 32 * 4);
    float* bnsum  = (float*)alloc(32 * 4);
    float* bnsq   = (float*)alloc(32 * 4);
    float* cnt    = (float*)alloc((size_t)NNODES * 4);
    float* pooled = (float*)alloc((size_t)N_GRAPHS * 32 * 4);
    float* gcnt   = (float*)alloc((size_t)N_GRAPHS * 4);
    float* ybuf   = (float*)alloc((size_t)NNODES * 32 * 4);
    float* feat   = (float*)alloc((size_t)NNODES * 32 * 4);
    _Float16* w1B0   = (_Float16*)alloc((size_t)4 * 32 * 16 * 2);
    _Float16* w1B1   = (_Float16*)alloc((size_t)4 * 32 * 16 * 2);
    _Float16* w2B0   = (_Float16*)alloc((size_t)2 * 64 * 32 * 16 * 2);
    _Float16* w2B1   = (_Float16*)alloc((size_t)2 * 64 * 32 * 16 * 2);
    _Float16* b2B0   = (_Float16*)alloc((size_t)2 * 32 * 16 * 2);
    _Float16* b2B1   = (_Float16*)alloc((size_t)2 * 32 * 16 * 2);
    _Float16* rootB0 = (_Float16*)alloc((size_t)2 * 32 * 16 * 2);
    _Float16* rootB1 = (_Float16*)alloc((size_t)2 * 32 * 16 * 2);

    // ---- zero accumulators ----
    zero_kernel<<<(NNODES * 32 + 255) / 256, 256, 0, stream>>>(agg, NNODES * 32);
    zero_kernel<<<1, 64, 0, stream>>>(bnsum, 32);
    zero_kernel<<<1, 64, 0, stream>>>(bnsq, 32);
    zero_kernel<<<(NNODES + 255) / 256, 256, 0, stream>>>(cnt, NNODES);
    zero_kernel<<<(N_GRAPHS * 32 + 255) / 256, 256, 0, stream>>>(pooled, N_GRAPHS * 32);
    zero_kernel<<<1, 64, 0, stream>>>(gcnt, N_GRAPHS);

    // ---- pack all B operands into WMMA fragment layout (f16) ----
    pack_b_kernel<<<1, 128, 0, stream>>>(w1_0, w1B0, 16, 64, 1, 4);
    pack_b_kernel<<<1, 128, 0, stream>>>(w1_1, w1B1, 16, 64, 1, 4);
    pack_b_kernel<<<16, 256, 0, stream>>>(w2_0, w2B0, 2048, 32, 64, 2); // w2 viewed [2048,32]
    pack_b_kernel<<<16, 256, 0, stream>>>(w2_1, w2B1, 2048, 32, 64, 2);
    pack_b_kernel<<<1, 64, 0, stream>>>(b2_0, b2B0, 32, 32, 1, 2);       // b2 viewed [32,32]
    pack_b_kernel<<<1, 64, 0, stream>>>(b2_1, b2B1, 32, 32, 1, 2);
    pack_b_kernel<<<1, 64, 0, stream>>>(root_0, rootB0, 32, 32, 1, 2);
    pack_b_kernel<<<1, 64, 0, stream>>>(root_1, rootB1, 32, 32, 1, 2);

    degree_kernel<<<(NEDGES + 255) / 256, 256, 0, stream>>>(ei, batch, cnt, gcnt);

    // ---- layer 0 ----
    nnconv_msg_kernel<<<NEDGES / 64, 128, 0, stream>>>(x, edge_attr, ei, w1B0, b1_0, w2B0, b2B0, agg);
    node_update_kernel<<<NNODES / 64, 128, 0, stream>>>(x, rootB0, bias_0, agg, cnt, ybuf, bnsum, bnsq);
    bn_relu_kernel<<<(NNODES * 32 + 255) / 256, 256, 0, stream>>>(ybuf, bnsum, bnsq, gamma_0, beta_0, feat);

    // ---- layer 1 ----
    zero_kernel<<<(NNODES * 32 + 255) / 256, 256, 0, stream>>>(agg, NNODES * 32);
    zero_kernel<<<1, 64, 0, stream>>>(bnsum, 32);
    zero_kernel<<<1, 64, 0, stream>>>(bnsq, 32);
    nnconv_msg_kernel<<<NEDGES / 64, 128, 0, stream>>>(feat, edge_attr, ei, w1B1, b1_1, w2B1, b2B1, agg);
    node_update_kernel<<<NNODES / 64, 128, 0, stream>>>(feat, rootB1, bias_1, agg, cnt, ybuf, bnsum, bnsq);
    bn_relu_kernel<<<(NNODES * 32 + 255) / 256, 256, 0, stream>>>(ybuf, bnsum, bnsq, gamma_1, beta_1, feat);

    // ---- pooling + final MLP ----
    pool_kernel<<<(NNODES * 32 + 255) / 256, 256, 0, stream>>>(feat, batch, pooled);
    final_mlp_kernel<<<1, 256, 0, stream>>>(pooled, gcnt, mlp_w1, mlp_b1, mlp_w2, mlp_b2, out);
}
